// MultiheadAttention_18769007084333
// MI455X (gfx1250) — matrix operand order repaired and checked by reading, not compiled, not run on hardware
//
#include <hip/hip_runtime.h>
#include <hip/hip_bf16.h>

#define BS    4
#define NCTX  2048
#define WIDTH 1024
#define HEADS 16
#define HD    64
#define QKV_N 3072
// 1/sqrt(sqrt(64))
#define ATTN_SCALE 0.35355339059327373f
#define LOG2E      1.4426950408889634f

typedef __attribute__((ext_vector_type(16))) __bf16 v16bf;
typedef __attribute__((ext_vector_type(2)))  __bf16 v2bf;
typedef __attribute__((ext_vector_type(8)))  float  v8f;

union AFrag { v16bf v; uint4 q[2]; unsigned short u[16]; };

// f32 -> bf16 via compiler fptrunc (expected to hit native v_cvt on gfx1250)
__device__ __forceinline__ unsigned short f2bf(float f) {
    __bf16 h = (__bf16)f;
    return __builtin_bit_cast(unsigned short, h);
}
__device__ __forceinline__ unsigned int pack2bf(float lo, float hi) {
    v2bf p = { (__bf16)lo, (__bf16)hi };
    return __builtin_bit_cast(unsigned int, p);
}

__device__ __forceinline__ float fast_exp2(float x) {
    return __builtin_amdgcn_exp2f(x);         // bare v_exp_f32
}
__device__ __forceinline__ int rfl(int v) {   // force wave-uniform into SGPR
    return __builtin_amdgcn_readfirstlane(v);
}

// ---------------------------------------------------------------------------
// Kernel 1: qkv = x @ w_qkv + b_qkv ; scatter to q/k bf16 [b,h,t,64] and
// v bf16 head-transposed [b,h,64,t]. Q additionally scaled by log2(e) so the
// attention kernel can use exp2. grid (24, 128), block 256, K-step 64.
// ---------------------------------------------------------------------------
__global__ __launch_bounds__(256) void mha_qkv_gemm(
    const float* __restrict__ x, const float* __restrict__ w,
    const float* __restrict__ bias,
    unsigned short* __restrict__ qb, unsigned short* __restrict__ kb,
    unsigned short* __restrict__ vb)
{
    __shared__ alignas(16) unsigned short lA[64 * 64];     // [m][k]  8KB
    __shared__ alignas(16) unsigned short lBt[128 * 64];   // [n][k] 16KB
    unsigned int* lA32  = (unsigned int*)lA;
    unsigned int* lBt32 = (unsigned int*)lBt;

    const int tid  = threadIdx.x;
    const int wave = rfl(tid >> 5);              // scalar wave id
    const int lane = tid & 31;
    const int half = lane >> 4, l = lane & 15;
    const int wm = wave & 3, wn = wave >> 2;     // 4 x 2 wave grid (scalar)
    const int m0 = blockIdx.y * 64;
    const int n0 = blockIdx.x * 128;

    v8f acc[4];
    v8f zv = {};
    #pragma unroll
    for (int i = 0; i < 4; ++i) acc[i] = zv;

    const int row = wm * 16 + l;

    for (int k0 = 0; k0 < WIDTH; k0 += 64) {
        { // stage A: 64x64 fp32 -> bf16 (packed dword writes)
            int m  = tid >> 2;
            int kk = (tid & 3) * 16;
            const float* src = x + (size_t)(m0 + m) * WIDTH + k0 + kk;
            #pragma unroll
            for (int j = 0; j < 8; ++j)
                lA32[(m * 64 + kk) / 2 + j] = pack2bf(src[2 * j], src[2 * j + 1]);
        }
        { // stage B transposed: lBt[n][k], (k,k+1) pairs per dword, 2 passes
            int nb = (tid & 15) * 8;
            #pragma unroll
            for (int kh = 0; kh < 2; ++kh) {
                int kp = (tid >> 4) + kh * 16;      // k-pair -> k = 2*kp
                const float* r0 = w + (size_t)(k0 + 2 * kp) * QKV_N + n0 + nb;
                const float* r1 = r0 + QKV_N;
                #pragma unroll
                for (int j = 0; j < 8; ++j)
                    lBt32[(nb + j) * 32 + kp] = pack2bf(r0[j], r1[j]);
            }
        }
        __syncthreads();

        #pragma unroll
        for (int s = 0; s < 2; ++s) {
            AFrag a;
            a.q[0] = *(const uint4*)&lA[row * 64 + 32 * s + 16 * half];
            a.q[1] = *(const uint4*)&lA[row * 64 + 32 * s + 16 * half + 8];
            #pragma unroll
            for (int nt = 0; nt < 4; ++nt) {
                AFrag bfr;
                int n = wn * 64 + nt * 16 + l;
                bfr.q[0] = *(const uint4*)&lBt[n * 64 + 32 * s + 16 * half];
                bfr.q[1] = *(const uint4*)&lBt[n * 64 + 32 * s + 16 * half + 8];
                acc[nt] = __builtin_amdgcn_wmma_f32_16x16x32_bf16(
                    false, a.v, false, bfr.v, (short)0, acc[nt], false, false);
            }
        }
        __syncthreads();
    }

    // Epilogue: h / typ / c-base and b / t-base are scalar (SALU) now.
    const int mb = m0 + wm * 16;
    const int bb = mb >> 11;
    const int tb = mb & (NCTX - 1);
    #pragma unroll
    for (int nt = 0; nt < 4; ++nt) {
        const int gbase = n0 + wn * 64 + nt * 16;   // scalar
        const int h   = gbase / 192;
        const int r   = gbase % 192;
        const int typ = r >> 6;
        const int cb  = r & 63;                     // c = cb + l
        const float bv = bias[gbase + l];
        if (typ == 2) {
            // v: [b,h,c,t]
            const size_t vbase =
                (((size_t)bb * HEADS + h) * HD + cb + l) * NCTX + tb;
            #pragma unroll
            for (int i = 0; i < 8; ++i)
                vb[vbase + i + half * 8] = f2bf(acc[nt][i] + bv);
        } else {
            unsigned short* dst = typ ? kb : qb;
            const float sc = typ ? ATTN_SCALE : (ATTN_SCALE * LOG2E);
            const size_t base =
                (((size_t)bb * HEADS + h) * NCTX + tb) * HD + cb;
            #pragma unroll
            for (int i = 0; i < 8; ++i)
                dst[base + (size_t)(i + half * 8) * HD + l] =
                    f2bf((acc[nt][i] + bv) * sc);
        }
    }
}

// ---------------------------------------------------------------------------
// Kernel 2: flash attention per (b,h); workgroup = 128 queries (8 waves x 16)
// grid (16, 16, 4), block 256. Streams 64-key K/V blocks via LDS.
// K in LDS as [key][c]; V in LDS as [c][key]. Softmax in the exp2 domain
// (log2(e) folded into Q by kernel 1).
// ---------------------------------------------------------------------------
__global__ __launch_bounds__(256) void mha_flash_attn(
    const unsigned short* __restrict__ qbuf,
    const unsigned short* __restrict__ kbuf,
    const unsigned short* __restrict__ vbuf,
    unsigned short* __restrict__ attn)
{
    __shared__ alignas(16) unsigned short lK[64 * 64];    // [key][c]  8KB
    __shared__ alignas(16) unsigned short lVt[64 * 64];   // [c][key]  8KB
    __shared__ alignas(16) unsigned short lP[8 * 16 * 64];// per-wave P 16KB

    const int tid  = threadIdx.x;
    const int wave = rfl(tid >> 5);
    const int lane = tid & 31;
    const int half = lane >> 4, l = lane & 15;
    const int b = blockIdx.z, h = blockIdx.y;
    const int qt0 = blockIdx.x * 128 + wave * 16;   // scalar

    const size_t head_base = ((size_t)(b * HEADS + h)) * NCTX * HD;

    // Q A-fragments (2 K-steps of 32 over head dim 64): contiguous b128 pairs
    AFrag aq[2];
    const unsigned short* qrow = qbuf + head_base + (size_t)(qt0 + l) * HD;
    #pragma unroll
    for (int s = 0; s < 2; ++s) {
        aq[s].q[0] = *(const uint4*)&qrow[32 * s + 16 * half];
        aq[s].q[1] = *(const uint4*)&qrow[32 * s + 16 * half + 8];
    }

    v8f accO[4];
    v8f zv = {};
    #pragma unroll
    for (int i = 0; i < 4; ++i) accO[i] = zv;
    float m_i[8], l_i[8];
    #pragma unroll
    for (int i = 0; i < 8; ++i) { m_i[i] = -1.0e30f; l_i[i] = 0.0f; }

    const uint4* kg4 = (const uint4*)(kbuf + head_base);   // [t][c] rows
    const uint4* vg4 = (const uint4*)(vbuf + head_base);   // [c][t] rows
    uint4* lK4 = (uint4*)lK;
    uint4* lV4 = (uint4*)lVt;
    const int prow = (wave * 16 + l) * 64;

    for (int kb0 = 0; kb0 < NCTX; kb0 += 64) {
        __syncthreads();  // prior iteration done with lVt / lP
        // K block (contiguous) and V block ([c][t] rows): 512 uint4 each
        for (int e = tid; e < 512; e += 256) {
            lK4[e] = kg4[kb0 * 8 + e];
            int c = e >> 3, wq = e & 7;
            lV4[e] = vg4[c * (NCTX / 8) + kb0 / 8 + wq];
        }
        __syncthreads();

        // S = Q @ K^T  (16 x 64 tile per wave), S already in log2 domain
        v8f sacc[4];
        #pragma unroll
        for (int i = 0; i < 4; ++i) sacc[i] = zv;
        #pragma unroll
        for (int nt = 0; nt < 4; ++nt) {
            int key = nt * 16 + l;
            #pragma unroll
            for (int s = 0; s < 2; ++s) {
                AFrag bfr;
                bfr.q[0] = *(const uint4*)&lK[key * 64 + 32 * s + 16 * half];
                bfr.q[1] = *(const uint4*)&lK[key * 64 + 32 * s + 16 * half + 8];
                sacc[nt] = __builtin_amdgcn_wmma_f32_16x16x32_bf16(
                    false, aq[s].v, false, bfr.v, (short)0, sacc[nt], false, false);
            }
        }

        // Online softmax update, exp2 domain (rows across 16 lanes of a half)
        float mnew[8], alpha[8];
        #pragma unroll
        for (int i = 0; i < 8; ++i) {
            float mx = fmaxf(fmaxf(sacc[0][i], sacc[1][i]),
                             fmaxf(sacc[2][i], sacc[3][i]));
            #pragma unroll
            for (int msk = 1; msk <= 8; msk <<= 1)
                mx = fmaxf(mx, __shfl_xor(mx, msk, 32));
            mnew[i]  = fmaxf(m_i[i], mx);
            alpha[i] = fast_exp2(m_i[i] - mnew[i]);
        }
        #pragma unroll
        for (int nt = 0; nt < 4; ++nt)
            #pragma unroll
            for (int i = 0; i < 8; ++i)
                sacc[nt][i] = fast_exp2(sacc[nt][i] - mnew[i]);
        #pragma unroll
        for (int i = 0; i < 8; ++i) {
            float rs = sacc[0][i] + sacc[1][i] + sacc[2][i] + sacc[3][i];
            #pragma unroll
            for (int msk = 1; msk <= 8; msk <<= 1)
                rs += __shfl_xor(rs, msk, 32);
            l_i[i] = l_i[i] * alpha[i] + rs;
            m_i[i] = mnew[i];
        }
        #pragma unroll
        for (int ct = 0; ct < 4; ++ct)
            #pragma unroll
            for (int i = 0; i < 8; ++i)
                accO[ct][i] *= alpha[i];

        // D-layout -> LDS (row m = i + 8*half, col = key)
        #pragma unroll
        for (int nt = 0; nt < 4; ++nt)
            #pragma unroll
            for (int i = 0; i < 8; ++i)
                lP[(wave * 16 + i + half * 8) * 64 + nt * 16 + l] =
                    f2bf(sacc[nt][i]);
        __syncthreads();

        // O += P @ V
        #pragma unroll
        for (int s = 0; s < 2; ++s) {
            AFrag pa;
            pa.q[0] = *(const uint4*)&lP[prow + 32 * s + 16 * half];
            pa.q[1] = *(const uint4*)&lP[prow + 32 * s + 16 * half + 8];
            #pragma unroll
            for (int ct = 0; ct < 4; ++ct) {
                int c = ct * 16 + l;
                AFrag bfr;
                bfr.q[0] = *(const uint4*)&lVt[c * 64 + 32 * s + 16 * half];
                bfr.q[1] = *(const uint4*)&lVt[c * 64 + 32 * s + 16 * half + 8];
                accO[ct] = __builtin_amdgcn_wmma_f32_16x16x32_bf16(
                    false, pa.v, false, bfr.v, (short)0, accO[ct], false, false);
            }
        }
    }

    // Normalize and write [b, t, h*64 + c] bf16 for the projection GEMM
    float inv[8];
    #pragma unroll
    for (int i = 0; i < 8; ++i) inv[i] = 1.0f / l_i[i];
    #pragma unroll
    for (int ct = 0; ct < 4; ++ct)
        #pragma unroll
        for (int i = 0; i < 8; ++i) {
            int t = qt0 + i + half * 8;
            int c = ct * 16 + l;
            attn[((size_t)b * NCTX + t) * WIDTH + h * HD + c] =
                f2bf(accO[ct][i] * inv[i]);
        }
}

// ---------------------------------------------------------------------------
// Kernel 3: out = attn @ w_proj + b_proj (fp32 out). grid (8, 128), block 256,
// K-step 64.
// ---------------------------------------------------------------------------
__global__ __launch_bounds__(256) void mha_proj_gemm(
    const unsigned short* __restrict__ a, const float* __restrict__ w,
    const float* __restrict__ bias, float* __restrict__ out)
{
    __shared__ alignas(16) unsigned short lA[64 * 64];     // [m][k]  8KB
    __shared__ alignas(16) unsigned short lBt[128 * 64];   // [n][k] 16KB
    unsigned int* lBt32 = (unsigned int*)lBt;

    const int tid  = threadIdx.x;
    const int wave = rfl(tid >> 5);
    const int lane = tid & 31;
    const int half = lane >> 4, l = lane & 15;
    const int wm = wave & 3, wn = wave >> 2;
    const int m0 = blockIdx.y * 64;
    const int n0 = blockIdx.x * 128;

    v8f acc[4];
    v8f zv = {};
    #pragma unroll
    for (int i = 0; i < 4; ++i) acc[i] = zv;

    const int row = wm * 16 + l;

    for (int k0 = 0; k0 < WIDTH; k0 += 64) {
        { // stage A (already bf16): two uint4 copies
            int m  = tid >> 2;
            int kk = (tid & 3) * 16;
            const unsigned short* src = a + (size_t)(m0 + m) * WIDTH + k0 + kk;
            *(uint4*)&lA[m * 64 + kk]     = *(const uint4*)(src);
            *(uint4*)&lA[m * 64 + kk + 8] = *(const uint4*)(src + 8);
        }
        { // stage B transposed fp32 -> bf16, 2 passes
            int nb = (tid & 15) * 8;
            #pragma unroll
            for (int kh = 0; kh < 2; ++kh) {
                int kp = (tid >> 4) + kh * 16;
                const float* r0 = w + (size_t)(k0 + 2 * kp) * WIDTH + n0 + nb;
                const float* r1 = r0 + WIDTH;
                #pragma unroll
                for (int j = 0; j < 8; ++j)
                    lBt32[(nb + j) * 32 + kp] = pack2bf(r0[j], r1[j]);
            }
        }
        __syncthreads();

        #pragma unroll
        for (int s = 0; s < 2; ++s) {
            AFrag afr;
            afr.q[0] = *(const uint4*)&lA[row * 64 + 32 * s + 16 * half];
            afr.q[1] = *(const uint4*)&lA[row * 64 + 32 * s + 16 * half + 8];
            #pragma unroll
            for (int nt = 0; nt < 4; ++nt) {
                AFrag bfr;
                int n = wn * 64 + nt * 16 + l;
                bfr.q[0] = *(const uint4*)&lBt[n * 64 + 32 * s + 16 * half];
                bfr.q[1] = *(const uint4*)&lBt[n * 64 + 32 * s + 16 * half + 8];
                acc[nt] = __builtin_amdgcn_wmma_f32_16x16x32_bf16(
                    false, afr.v, false, bfr.v, (short)0, acc[nt], false, false);
            }
        }
        __syncthreads();
    }

    #pragma unroll
    for (int nt = 0; nt < 4; ++nt) {
        int gn = n0 + wn * 64 + nt * 16 + l;
        float bv = bias[gn];
        #pragma unroll
        for (int i = 0; i < 8; ++i) {
            int gm = m0 + wm * 16 + i + half * 8;
            out[(size_t)gm * WIDTH + gn] = acc[nt][i] + bv;
        }
    }
}

// ---------------------------------------------------------------------------
extern "C" void kernel_launch(void* const* d_in, const int* in_sizes, int n_in,
                              void* d_out, int out_size, void* d_ws, size_t ws_size,
                              hipStream_t stream)
{
    const float* x      = (const float*)d_in[0];
    const float* w_qkv  = (const float*)d_in[1];
    const float* b_qkv  = (const float*)d_in[2];
    const float* w_proj = (const float*)d_in[3];
    const float* b_proj = (const float*)d_in[4];
    float* out = (float*)d_out;

    char* ws = (char*)d_ws;
    const size_t HBUF = (size_t)BS * HEADS * NCTX * HD * sizeof(unsigned short); // 16 MB
    unsigned short* qb   = (unsigned short*)(ws);
    unsigned short* kb   = (unsigned short*)(ws + HBUF);
    unsigned short* vb   = (unsigned short*)(ws + 2 * HBUF);
    unsigned short* attn = (unsigned short*)(ws + 3 * HBUF);

    mha_qkv_gemm<<<dim3(QKV_N / 128, (BS * NCTX) / 64), 256, 0, stream>>>(
        x, w_qkv, b_qkv, qb, kb, vb);
    mha_flash_attn<<<dim3(NCTX / 128, HEADS, BS), 256, 0, stream>>>(
        qb, kb, vb, attn);
    mha_proj_gemm<<<dim3(WIDTH / 128, (BS * NCTX) / 64), 256, 0, stream>>>(
        attn, w_proj, b_proj, out);
}
